// GNN_inv_HierarchicalAttention_60687887892633
// MI455X (gfx1250) — matrix operand order repaired and checked by reading, not compiled, not run on hardware
//
#include <hip/hip_runtime.h>
#include <hip/hip_bf16.h>
#include <math.h>

// ---------------------------------------------------------------------------
// GNN hierarchical-attention forward for MI455X (gfx1250).
//  * Dense X@W -> v_wmma_f32_16x16x32_f16, 64x32 tile per wave (8 WMMA/K-step),
//    branch-free clamped loads; K and N compile-time so B-loads/C-stores use
//    immediate offsets; wave-uniform (readfirstlane) tile indices so tile
//    selection / epilogue guards are scalar branches, not EXEC masks.
//  * Edge-encoder front fused (proc tile stays in LDS/registers).
//  * Segment softmax / mean / max via fp32 global atomics.
// ---------------------------------------------------------------------------

#define NN 25000      // N_NODES
#define NE 400000     // N_EDGES
#define EAF (NE + NN) // edges + self loops

typedef __attribute__((ext_vector_type(16))) _Float16 v16h;
typedef __attribute__((ext_vector_type(8)))  float    v8f;

__device__ __forceinline__ float wave_sum32(float v) {
#pragma unroll
  for (int off = 16; off > 0; off >>= 1) v += __shfl_xor(v, off, 32);
  return v;
}

__device__ __forceinline__ void atomicMaxF(float* addr, float val) {
  unsigned int* ua = (unsigned int*)addr;
  unsigned int cur = __float_as_uint(*addr);
  while (__uint_as_float(cur) < val) {
    unsigned int prev = atomicCAS(ua, cur, __float_as_uint(val));
    if (prev == cur) break;
    cur = prev;
  }
}

__device__ __forceinline__ int imin(int a, int b) { return a < b ? a : b; }

// wave id as a compiler-provable scalar
__device__ __forceinline__ int wave_id() {
  return __builtin_amdgcn_readfirstlane((int)(threadIdx.x >> 5));
}

// ---------------------------------------------------------------------------
// WMMA GEMM: C[M x N] = act(A[M x K] @ W[K x N] + bias). fp32 memory,
// f16 operands / f32 accumulate. One wave computes a 64x32 output tile:
// 4 A-fragments x 2 B-fragments -> 8 v_wmma per 32-wide K step.
// ACT: 0 = identity, 1 = relu. K, N compile-time.
// ---------------------------------------------------------------------------
template <int ACT, int K, int N>
__global__ void k_gemm(const float* __restrict__ A, const float* __restrict__ W,
                       const float* __restrict__ bias, float* __restrict__ C,
                       int M) {
  constexpr int tnT = N >> 5;                                // 32-col tiles
  const int tile = blockIdx.x * (blockDim.x >> 5) + wave_id(); // scalar
  const int tm   = tile / tnT;
  const int tn   = tile - tm * tnT;
  if (tm * 64 >= M) return;                                  // scalar branch

  const int lane = threadIdx.x & 31;
  const int l15  = lane & 15;
  const int hi   = lane >> 4;        // 0: lanes 0-15, 1: lanes 16-31
  const int aKhi = hi ? 8 : 0;       // A frag: upper half-wave -> K+8 block
  const int bKhi = hi ? 8 : 0;       // B frag: upper half-wave -> K+8 block

  // Clamped A-row base pointers (out-of-range rows read row M-1; never stored)
  const float* Arow[4];
#pragma unroll
  for (int i = 0; i < 4; ++i) {
    const int r = imin(tm * 64 + i * 16 + l15, M - 1);
    Arow[i] = A + (size_t)r * K;
  }
  const int colB0 = tn * 32 + l15;   // B frag j uses column colB0 + 16*j
  const float* Wc0 = W + colB0;      // immediate offsets: + k*N + 16*j

  v8f acc[4][2];
#pragma unroll
  for (int i = 0; i < 4; ++i)
#pragma unroll
    for (int j = 0; j < 2; ++j) acc[i][j] = v8f{0.f,0.f,0.f,0.f,0.f,0.f,0.f,0.f};

  constexpr int KF = (K / 32) * 32;

  for (int k0 = 0; k0 < KF; k0 += 32) {
    if (k0 + 32 < KF) {                                      // scalar branch
#pragma unroll
      for (int i = 0; i < 4; ++i)
        __builtin_prefetch(Arow[i] + k0 + 32, 0, 1);
    }
    v16h a[4], b[2];
#pragma unroll
    for (int i = 0; i < 4; ++i) {
#pragma unroll
      for (int v = 0; v < 8; ++v) {
        const int ka = k0 + ((v >= 4) ? 16 : 0) + aKhi + ((v & 3) << 1);
        if constexpr ((K & 1) == 0) {                        // 8B-aligned pair
          const float2 t = *(const float2*)(Arow[i] + ka);
          a[i][2 * v]     = (_Float16)t.x;
          a[i][2 * v + 1] = (_Float16)t.y;
        } else {
          a[i][2 * v]     = (_Float16)Arow[i][ka];
          a[i][2 * v + 1] = (_Float16)Arow[i][ka + 1];
        }
      }
    }
#pragma unroll
    for (int j = 0; j < 2; ++j) {
#pragma unroll
      for (int v = 0; v < 8; ++v) {
        const int kb = k0 + ((v >= 4) ? 16 : 0) + bKhi + ((v & 3) << 1);
        b[j][2 * v]     = (_Float16)Wc0[(size_t)kb * N + 16 * j];
        b[j][2 * v + 1] = (_Float16)Wc0[(size_t)(kb + 1) * N + 16 * j];
      }
    }
#pragma unroll
    for (int i = 0; i < 4; ++i)
#pragma unroll
      for (int j = 0; j < 2; ++j)
        acc[i][j] = __builtin_amdgcn_wmma_f32_16x16x32_f16(
            false, a[i], false, b[j], (short)0, acc[i][j], false, false);
  }

  if constexpr (KF < K) {  // K tail: clamped addresses + value selects
    v16h a[4], b[2];
#pragma unroll
    for (int i = 0; i < 4; ++i) {
#pragma unroll
      for (int v = 0; v < 8; ++v) {
        const int ka = KF + ((v >= 4) ? 16 : 0) + aKhi + ((v & 3) << 1);
        const float a0 = Arow[i][imin(ka, K - 1)];
        const float a1 = Arow[i][imin(ka + 1, K - 1)];
        a[i][2 * v]     = (_Float16)(ka     < K ? a0 : 0.f);
        a[i][2 * v + 1] = (_Float16)(ka + 1 < K ? a1 : 0.f);
      }
    }
#pragma unroll
    for (int j = 0; j < 2; ++j) {
#pragma unroll
      for (int v = 0; v < 8; ++v) {
        const int kb = KF + ((v >= 4) ? 16 : 0) + bKhi + ((v & 3) << 1);
        const float b0 = Wc0[(size_t)imin(kb, K - 1) * N + 16 * j];
        const float b1 = Wc0[(size_t)imin(kb + 1, K - 1) * N + 16 * j];
        b[j][2 * v]     = (_Float16)(kb     < K ? b0 : 0.f);
        b[j][2 * v + 1] = (_Float16)(kb + 1 < K ? b1 : 0.f);
      }
    }
#pragma unroll
    for (int i = 0; i < 4; ++i)
#pragma unroll
      for (int j = 0; j < 2; ++j)
        acc[i][j] = __builtin_amdgcn_wmma_f32_16x16x32_f16(
            false, a[i], false, b[j], (short)0, acc[i][j], false, false);
  }

  // Epilogue. C/D layout: lane col = lane&15, VGPR r -> row r + 8*(lane>=16)
  const bool full = (tm * 64 + 64) <= M;                     // scalar
#pragma unroll
  for (int j = 0; j < 2; ++j) {
    const int ccol = tn * 32 + 16 * j + l15;
    const float bv = bias ? bias[ccol] : 0.f;
#pragma unroll
    for (int i = 0; i < 4; ++i) {
      const int rbase = tm * 64 + 16 * i + (hi ? 8 : 0);
      float* cp = C + (size_t)rbase * N + ccol;
      if (full) {
#pragma unroll
        for (int r = 0; r < 8; ++r) {
          float v = acc[i][j][r] + bv;
          if (ACT == 1) v = fmaxf(v, 0.f);
          cp[r * N] = v;                                     // immediate offsets
        }
      } else {
#pragma unroll
        for (int r = 0; r < 8; ++r) {
          if (rbase + r < M) {
            float v = acc[i][j][r] + bv;
            if (ACT == 1) v = fmaxf(v, 0.f);
            cp[r * N] = v;
          }
        }
      }
    }
  }
}

// ---------------------------------------------------------------------------
// Fused edge-encoder front: per edge, freq MLP (2->64, relu), attention
// (64->32 tanh, 32->1), softmax over 8 freqs, weighted sum -> dyn[64];
// writes comb[e] = [dyn(64) | static(3)]. One wave per edge; proc in LDS.
// ---------------------------------------------------------------------------
__global__ void k_edge_front(const float* __restrict__ x_edges,
                             const float* __restrict__ fpw, const float* __restrict__ fpb,
                             const float* __restrict__ a1w, const float* __restrict__ a1b,
                             const float* __restrict__ a2w, const float* __restrict__ a2b,
                             float* __restrict__ comb) {
  __shared__ float sproc[8][8][64]; // [wave][freq][chan]
  const int lane = threadIdx.x & 31;
  const int wid  = wave_id();
  const int e    = blockIdx.x * 8 + wid;       // scalar
  if (e >= NE) return; // NE % 8 == 0: no wave diverges at the barrier

  const float* row = x_edges + (size_t)e * 19;
  const float w0a = fpw[lane],      w0b = fpw[lane + 32];
  const float w1a = fpw[64 + lane], w1b = fpw[64 + lane + 32];
  const float ba  = fpb[lane],      bb  = fpb[lane + 32];

  float p[8][2];
#pragma unroll
  for (int q = 0; q < 8; ++q) {
    const float f0 = row[3 + 2 * q], f1 = row[3 + 2 * q + 1];
    const float va = fmaxf(f0 * w0a + f1 * w1a + ba, 0.f);
    const float vb = fmaxf(f0 * w0b + f1 * w1b + bb, 0.f);
    p[q][0] = va; p[q][1] = vb;
    sproc[wid][q][lane]      = va;
    sproc[wid][q][lane + 32] = vb;
  }
  __syncthreads();

  const float a2v = a2w[lane];
  const float a1bias = a1b[lane];
  float s[8];
#pragma unroll
  for (int q = 0; q < 8; ++q) {
    float t = a1bias;
    for (int c = 0; c < 64; ++c) t += sproc[wid][q][c] * a1w[c * 32 + lane];
    t = tanhf(t);
    s[q] = wave_sum32(t * a2v) + a2b[0]; // butterfly: all lanes get s[q]
  }
  float mx = s[0];
#pragma unroll
  for (int q = 1; q < 8; ++q) mx = fmaxf(mx, s[q]);
  float den = 0.f;
#pragma unroll
  for (int q = 0; q < 8; ++q) { s[q] = expf(s[q] - mx); den += s[q]; }
  const float inv = 1.f / den;

  float dyn0 = 0.f, dyn1 = 0.f;
#pragma unroll
  for (int q = 0; q < 8; ++q) {
    const float wq = s[q] * inv;
    dyn0 += wq * p[q][0];
    dyn1 += wq * p[q][1];
  }
  float* crow = comb + (size_t)e * 67;
  crow[lane]      = dyn0;
  crow[lane + 32] = dyn1;
  if (lane < 3) crow[64 + lane] = row[lane]; // static features
}

// ---------------------------------------------------------------------------
__global__ void k_fill(float* __restrict__ p, float v, long long n) {
  const long long i = (long long)blockIdx.x * blockDim.x + threadIdx.x;
  if (i < n) p[i] = v;
}

__global__ void k_msg_scatter(const float* __restrict__ emb, const int* __restrict__ ei,
                              float* __restrict__ msg, float* __restrict__ cnt) {
  const long long i = (long long)blockIdx.x * blockDim.x + threadIdx.x;
  if (i >= (long long)NE * 64) return;
  const int e = (int)(i >> 6);
  const int f = (int)(i & 63);
  const int d = ei[NE + e];
  atomicAdd(&msg[(size_t)d * 64 + f], emb[i]);
  if (f == 0) atomicAdd(&cnt[d], 1.0f);
}

__global__ void k_combine(const float* __restrict__ nemb, const float* __restrict__ msg,
                          const float* __restrict__ cnt, float* __restrict__ x0) {
  const long long i = (long long)blockIdx.x * blockDim.x + threadIdx.x;
  if (i >= (long long)NN * 64) return;
  const int n = (int)(i >> 6);
  x0[i] = nemb[i] + msg[i] / fmaxf(cnt[n], 1.0f);
}

__global__ void k_attn_coef(const float* __restrict__ h,
                            const float* __restrict__ ws, const float* __restrict__ wd,
                            float* __restrict__ as, float* __restrict__ ad, int H) {
  const int i = blockIdx.x * blockDim.x + threadIdx.x;
  if (i >= NN * H) return;
  const int n = i / H, hh = i - n * H;
  const float* hr = h + (size_t)n * H * 64 + (size_t)hh * 64;
  float s1 = 0.f, s2 = 0.f;
  for (int d = 0; d < 64; ++d) { const float v = hr[d]; s1 += v * ws[hh * 64 + d]; s2 += v * wd[hh * 64 + d]; }
  as[i] = s1; ad[i] = s2;
}

__device__ __forceinline__ void edge_sd(const int* ei, int e, int& s, int& d) {
  if (e < NE) { s = ei[e]; d = ei[NE + e]; } else { s = d = e - NE; }
}

__global__ void k_edge_max(const int* __restrict__ ei, const float* __restrict__ as,
                           const float* __restrict__ ad, float* __restrict__ eb,
                           float* __restrict__ m, int H) {
  const long long i = (long long)blockIdx.x * blockDim.x + threadIdx.x;
  if (i >= (long long)EAF * H) return;
  const int e = (int)(i / H), hh = (int)(i - (long long)e * H);
  int s, d; edge_sd(ei, e, s, d);
  float v = as[s * H + hh] + ad[d * H + hh];
  v = v > 0.f ? v : 0.2f * v; // leaky relu
  eb[i] = v;
  atomicMaxF(&m[d * H + hh], v);
}

__global__ void k_edge_expsum(const int* __restrict__ ei, float* __restrict__ eb,
                              const float* __restrict__ m, float* __restrict__ den, int H) {
  const long long i = (long long)blockIdx.x * blockDim.x + threadIdx.x;
  if (i >= (long long)EAF * H) return;
  const int e = (int)(i / H), hh = (int)(i - (long long)e * H);
  int s, d; edge_sd(ei, e, s, d);
  const float ex = expf(eb[i] - m[d * H + hh]);
  eb[i] = ex;
  atomicAdd(&den[d * H + hh], ex);
}

__global__ void k_edge_agg(const int* __restrict__ ei, const float* __restrict__ eb,
                           const float* __restrict__ den, const float* __restrict__ h,
                           float* __restrict__ out, int H) {
  const long long i = (long long)blockIdx.x * blockDim.x + threadIdx.x;
  if (i >= (long long)EAF * H * 64) return;
  const int f = (int)(i & 63);
  const long long eh = i >> 6;
  const int e = (int)(eh / H), hh = (int)(eh - (long long)e * H);
  int s, d; edge_sd(ei, e, s, d);
  const float alpha = eb[eh] / den[d * H + hh];
  atomicAdd(&out[((size_t)d * H + hh) * 64 + f],
            alpha * h[((size_t)s * H + hh) * 64 + f]);
}

// bias + layernorm + elu, one wave per node, C in {64, 256}
__global__ void k_post(const float* __restrict__ agg, const float* __restrict__ bias,
                       const float* __restrict__ g, const float* __restrict__ b,
                       float* __restrict__ out, int C) {
  const int lane = threadIdx.x & 31;
  const int node = blockIdx.x * (blockDim.x >> 5) + wave_id(); // scalar
  if (node >= NN) return;
  const float* row = agg + (size_t)node * C;
  const int per = C >> 5; // 2 or 8
  float v[8];
  float sum = 0.f;
  for (int i = 0; i < per; ++i) { const int c = lane + (i << 5); v[i] = row[c] + bias[c]; sum += v[i]; }
  const float mu = wave_sum32(sum) / (float)C;
  float var = 0.f;
  for (int i = 0; i < per; ++i) { const float d = v[i] - mu; var += d * d; }
  var = wave_sum32(var) / (float)C;
  const float rs = rsqrtf(var + 1e-5f);
  float* orow = out + (size_t)node * C;
  for (int i = 0; i < per; ++i) {
    const int c = lane + (i << 5);
    float y = (v[i] - mu) * rs * g[c] + b[c];
    orow[c] = y > 0.f ? y : expm1f(y); // elu
  }
}

__global__ void k_pool_max(const float* __restrict__ x, float* __restrict__ pooled) {
  const long long i = (long long)blockIdx.x * blockDim.x + threadIdx.x;
  if (i >= (long long)NN * 64) return;
  atomicMaxF(&pooled[i & 63], x[i]);
}

__global__ void k_decoder(const float* __restrict__ pooled,
                          const float* __restrict__ w1, const float* __restrict__ b1,
                          const float* __restrict__ w2, const float* __restrict__ b2,
                          const float* __restrict__ w3, const float* __restrict__ b3,
                          float* __restrict__ out) {
  __shared__ float pl[64];
  __shared__ float h1[256];
  __shared__ float h2[256];
  const int t = threadIdx.x;
  if (t < 64) pl[t] = pooled[t];
  __syncthreads();
  float a = b1[t];
  for (int k = 0; k < 64; ++k) a += pl[k] * w1[k * 256 + t];
  h1[t] = fmaxf(a, 0.f);
  __syncthreads();
  float c = b2[t];
  for (int k = 0; k < 256; ++k) c += h1[k] * w2[k * 256 + t];
  h2[t] = fmaxf(c, 0.f);
  __syncthreads();
  if (t < 8) {
    float o = b3[t];
    for (int k = 0; k < 256; ++k) o += h2[k] * w3[k * 8 + t];
    out[t] = o;
  }
}

// ---------------------------------------------------------------------------
static inline int cdiv(long long a, long long b) { return (int)((a + b - 1) / b); }

// host-side dispatch to template instantiations
static void launch_gemm(int act, int K, const float* A, const float* W,
                        const float* bias, float* C, int M, int N,
                        hipStream_t stream) {
  const int TB = 256, WPB = TB / 32;
  const int tiles = cdiv(M, 64) * (N / 32);
  const int grid = cdiv(tiles, WPB);
  if      (act == 1 && K == 16  && N == 64)  k_gemm<1, 16,  64 ><<<grid, TB, 0, stream>>>(A, W, bias, C, M);
  else if (act == 0 && K == 64  && N == 64)  k_gemm<0, 64,  64 ><<<grid, TB, 0, stream>>>(A, W, bias, C, M);
  else if (act == 1 && K == 67  && N == 64)  k_gemm<1, 67,  64 ><<<grid, TB, 0, stream>>>(A, W, bias, C, M);
  else if (act == 0 && K == 64  && N == 256) k_gemm<0, 64,  256><<<grid, TB, 0, stream>>>(A, W, bias, C, M);
  else if (act == 0 && K == 256 && N == 256) k_gemm<0, 256, 256><<<grid, TB, 0, stream>>>(A, W, bias, C, M);
  else if (act == 0 && K == 256 && N == 64)  k_gemm<0, 256, 64 ><<<grid, TB, 0, stream>>>(A, W, bias, C, M);
}

extern "C" void kernel_launch(void* const* d_in, const int* in_sizes, int n_in,
                              void* d_out, int out_size, void* d_ws, size_t ws_size,
                              hipStream_t stream) {
  (void)in_sizes; (void)n_in; (void)out_size; (void)ws_size;

  const float* x_nodes = (const float*)d_in[0];
  const float* x_edges = (const float*)d_in[1];
  const int*   ei      = (const int*)d_in[2];
  // d_in[3] = batch (all zeros, single graph) — unused

  // params leaves in JAX pytree (sorted-key) order
  int p = 4;
  const float* dec_l1_b = (const float*)d_in[p++];
  const float* dec_l1_w = (const float*)d_in[p++];
  const float* dec_l2_b = (const float*)d_in[p++];
  const float* dec_l2_w = (const float*)d_in[p++];
  const float* dec_l3_b = (const float*)d_in[p++];
  const float* dec_l3_w = (const float*)d_in[p++];
  const float* att1_b = (const float*)d_in[p++];
  const float* att1_w = (const float*)d_in[p++];
  const float* att2_b = (const float*)d_in[p++];
  const float* att2_w = (const float*)d_in[p++];
  const float* cmb1_b = (const float*)d_in[p++];
  const float* cmb1_w = (const float*)d_in[p++];
  const float* cmb2_b = (const float*)d_in[p++];
  const float* cmb2_w = (const float*)d_in[p++];
  const float* fp_b   = (const float*)d_in[p++];
  const float* fp_w   = (const float*)d_in[p++];
  const float *g_ad[3], *g_as[3], *g_b[3], *g_lnb[3], *g_lng[3], *g_w[3];
  for (int l = 0; l < 3; ++l) {
    g_ad[l]  = (const float*)d_in[p++];
    g_as[l]  = (const float*)d_in[p++];
    g_b[l]   = (const float*)d_in[p++];
    g_lnb[l] = (const float*)d_in[p++];
    g_lng[l] = (const float*)d_in[p++];
    g_w[l]   = (const float*)d_in[p++];
  }
  const float* ne_l1_b = (const float*)d_in[p++];
  const float* ne_l1_w = (const float*)d_in[p++];
  const float* ne_l2_b = (const float*)d_in[p++];
  const float* ne_l2_w = (const float*)d_in[p++];

  // workspace carve (256B aligned)
  char* wsp = (char*)d_ws;
  auto carve = [&](size_t bytes) -> float* {
    float* r = (float*)wsp;
    wsp += (bytes + 255) & ~(size_t)255;
    return r;
  };
  float* comb     = carve((size_t)NE * 67 * 4);
  float* h1e      = carve((size_t)NE * 64 * 4);
  float* edge_emb = carve((size_t)NE * 64 * 4);
  float* node_h1  = carve((size_t)NN * 64 * 4);
  float* node_emb = carve((size_t)NN * 64 * 4);
  float* msg      = carve((size_t)NN * 64 * 4);
  float* cnt      = carve((size_t)NN * 4);
  float* xb0      = carve((size_t)NN * 256 * 4);
  float* xb1      = carve((size_t)NN * 256 * 4);
  float* hb       = carve((size_t)NN * 256 * 4);
  float* ob       = carve((size_t)NN * 256 * 4);
  float* asrc     = carve((size_t)NN * 4 * 4);
  float* adst     = carve((size_t)NN * 4 * 4);
  float* mbuf     = carve((size_t)NN * 4 * 4);
  float* dbuf     = carve((size_t)NN * 4 * 4);
  float* ebuf     = carve((size_t)EAF * 4 * 4);
  float* pooled   = carve(64 * 4);

  const int TB = 256, WPB = TB / 32;

  // ---- node encoder: 16->64 relu, 64->64 -------------------------------
  launch_gemm(1, 16, x_nodes, ne_l1_w, ne_l1_b, node_h1, NN, 64, stream);
  launch_gemm(0, 64, node_h1, ne_l2_w, ne_l2_b, node_emb, NN, 64, stream);

  // ---- edge encoder ----------------------------------------------------
  k_edge_front<<<cdiv(NE, 8), TB, 0, stream>>>(x_edges, fp_w, fp_b, att1_w, att1_b,
                                               att2_w, att2_b, comb);
  launch_gemm(1, 67, comb, cmb1_w, cmb1_b, h1e, NE, 64, stream);
  launch_gemm(0, 64, h1e, cmb2_w, cmb2_b, edge_emb, NE, 64, stream);

  // ---- message aggregation -> x0 --------------------------------------
  k_fill<<<cdiv((long long)NN * 64, TB), TB, 0, stream>>>(msg, 0.f, (long long)NN * 64);
  k_fill<<<cdiv((long long)NN, TB), TB, 0, stream>>>(cnt, 0.f, (long long)NN);
  k_msg_scatter<<<cdiv((long long)NE * 64, TB), TB, 0, stream>>>(edge_emb, ei, msg, cnt);
  k_combine<<<cdiv((long long)NN * 64, TB), TB, 0, stream>>>(node_emb, msg, cnt, xb0);

  // ---- 3 GAT layers ----------------------------------------------------
  const int Hs[3]  = {4, 4, 1};
  const int Kin[3] = {64, 256, 256};
  float* xin = xb0;
  float* xout = xb1;
  for (int l = 0; l < 3; ++l) {
    const int H = Hs[l], C = H * 64, K = Kin[l];
    launch_gemm(0, K, xin, g_w[l], nullptr, hb, NN, C, stream);
    k_attn_coef<<<cdiv((long long)NN * H, TB), TB, 0, stream>>>(hb, g_as[l], g_ad[l], asrc, adst, H);
    k_fill<<<cdiv((long long)NN * H, TB), TB, 0, stream>>>(mbuf, -1e30f, (long long)NN * H);
    k_fill<<<cdiv((long long)NN * H, TB), TB, 0, stream>>>(dbuf, 0.f, (long long)NN * H);
    k_fill<<<cdiv((long long)NN * C, TB), TB, 0, stream>>>(ob, 0.f, (long long)NN * C);
    k_edge_max<<<cdiv((long long)EAF * H, TB), TB, 0, stream>>>(ei, asrc, adst, ebuf, mbuf, H);
    k_edge_expsum<<<cdiv((long long)EAF * H, TB), TB, 0, stream>>>(ei, ebuf, mbuf, dbuf, H);
    k_edge_agg<<<cdiv((long long)EAF * H * 64, TB), TB, 0, stream>>>(ei, ebuf, dbuf, hb, ob, H);
    k_post<<<cdiv(NN, WPB), TB, 0, stream>>>(ob, g_b[l], g_lng[l], g_lnb[l], xout, C);
    float* t = xin; xin = xout; xout = t;
  }
  // final node features now in xin (64-dim rows)

  // ---- global max pool + decoder --------------------------------------
  k_fill<<<1, 64, 0, stream>>>(pooled, -1e30f, 64);
  k_pool_max<<<cdiv((long long)NN * 64, TB), TB, 0, stream>>>(xin, pooled);
  k_decoder<<<1, 256, 0, stream>>>(pooled, dec_l1_w, dec_l1_b, dec_l2_w, dec_l2_b,
                                   dec_l3_w, dec_l3_b, (float*)d_out);
}
